// CRHT_DGC_82489141887717
// MI455X (gfx1250) — matrix operand order repaired and checked by reading, not compiled, not run on hardware
//
#include <hip/hip_runtime.h>
#include <hip/hip_bf16.h>

// ---------------------------------------------------------------------------
// CTR-GCN block for MI455X (gfx1250, wave32, WMMA f32<-f16 16x16x32).
// All channel GEMMs + PA mixing + depthwise CTRGC einsum use
// v_wmma_f32_16x16x32_f16; BN folded into f16 weights. Fragment packing is
// done from b128-loaded register rows with constant-index selects.
// ---------------------------------------------------------------------------

typedef _Float16 half16 __attribute__((ext_vector_type(16)));
typedef float    f32x8  __attribute__((ext_vector_type(8)));

#define L_    3
#define S_    3
#define V_    25
#define CIN_  64
#define COUT_ 256
#define INTER_ 64
#define REL_  8
#define N_    32
#define T_    128
#define ROWS_ 102400   // N*T*V
#define EPS_  1e-5f

// ---- workspace layout (bytes) ----
constexpr size_t OFF_XK   = 0;                                   // f16 ROWS x 64
constexpr size_t OFF_XD   = OFF_XK   + (size_t)ROWS_*64*2;       // f16 L x ROWS x 64
constexpr size_t OFF_U    = OFF_XD   + (size_t)L_*ROWS_*64*2;    // f16 L x ROWS x 192 (subset convs)
constexpr size_t OFF_X3T  = OFF_U    + (size_t)L_*ROWS_*192*2;   // f16 L x N x 64 x (T*V)  (c3 conv, transposed)
constexpr size_t OFF_CBUF = OFF_X3T  + (size_t)L_*N_*64*T_*V_*2; // f32 N x 64 x T x V
constexpr size_t OFF_MBUF = OFF_CBUF + (size_t)N_*64*T_*V_*4;    // f16 L x N x 64 x 25 x 25
constexpr size_t OFF_XDM  = OFF_MBUF + (size_t)L_*N_*64*625*2;   // f32 L x N x 25 x 64
constexpr size_t OFF_CDW  = OFF_XDM  + (size_t)L_*N_*25*64*4;    // f16 L x 64 x 64
constexpr size_t OFF_UW   = OFF_CDW  + (size_t)L_*64*64*2;       // f16 L x 256 x 64
constexpr size_t OFF_DW16 = OFF_UW   + (size_t)L_*256*64*2;      // f16 256 x 64
constexpr size_t OFF_PA16 = OFF_DW16 + (size_t)256*64*2;         // f16 L x S x 32 x 32
constexpr size_t OFF_CDB  = OFF_PA16 + (size_t)L_*S_*32*32*2;    // f32 192
constexpr size_t OFF_C3B  = OFF_CDB  + 192*4;                    // f32 192
constexpr size_t OFF_BALL = OFF_C3B  + 192*4;                    // f32 256
// total ~244 MB

__device__ __forceinline__ f32x8 wmma16(half16 a, half16 b, f32x8 c) {
  return __builtin_amdgcn_wmma_f32_16x16x32_f16(false, a, false, b, (short)0, c,
                                                false, false);
}

// A fragment (16x32 f16): lane row M=lane&15; lanes<16 hold K=0..15,
// lanes>=16 hold K=16..31 (matches B). hr = 64 contiguous K values.
__device__ __forceinline__ void build_a(const _Float16* hr, int lhalf,
                                        half16& a0, half16& a1) {
#pragma unroll
  for (int vg = 0; vg < 8; ++vg)
#pragma unroll
    for (int e = 0; e < 2; ++e) {
      const int b0 = vg * 2 + e;
      a0[vg * 2 + e] = lhalf ? hr[b0 + 16] : hr[b0];
      a1[vg * 2 + e] = lhalf ? hr[b0 + 48] : hr[b0 + 32];
    }
}
// single-K-step variant, hr = 32 contiguous K values
__device__ __forceinline__ half16 build_a32(const _Float16* hr, int lhalf) {
  half16 a;
#pragma unroll
  for (int vg = 0; vg < 8; ++vg)
#pragma unroll
    for (int e = 0; e < 2; ++e) {
      const int b0 = vg * 2 + e;
      a[vg * 2 + e] = lhalf ? hr[b0 + 16] : hr[b0];
    }
  return a;
}
// B fragment (32x16 f16): lane col N=lane&15; lanes<16 hold K=0..15,
// lanes>=16 hold K=16..31. hr = 64 contiguous K values for this column.
__device__ __forceinline__ void build_b(const _Float16* hr, int lhalf,
                                        half16& b0v, half16& b1v) {
#pragma unroll
  for (int vg = 0; vg < 8; ++vg)
#pragma unroll
    for (int e = 0; e < 2; ++e) {
      const int k = vg * 2 + e;
      b0v[vg * 2 + e] = lhalf ? hr[k + 16] : hr[k];
      b1v[vg * 2 + e] = lhalf ? hr[k + 48] : hr[k + 32];
    }
}
__device__ __forceinline__ half16 build_b32(const _Float16* hr, int lhalf) {
  half16 b;
#pragma unroll
  for (int vg = 0; vg < 8; ++vg)
#pragma unroll
    for (int e = 0; e < 2; ++e) {
      const int k = vg * 2 + e;
      b[vg * 2 + e] = lhalf ? hr[k + 16] : hr[k];
    }
  return b;
}

// ---------------------------------------------------------------------------
// K0: fold BatchNorms into f16 weights / f32 biases, pad PA to 32x32 f16.
// ---------------------------------------------------------------------------
__global__ void k_prep(const float* PA, const float* cdw, const float* cdb,
                       const float* cdg, const float* cdbe, const float* cdm,
                       const float* cdv, const float* sw, const float* sb,
                       const float* sg, const float* sbe, const float* sm,
                       const float* sv, const float* c3w, const float* c3b,
                       const float* bng, const float* bnb, const float* bnm,
                       const float* bnv, const float* dw, const float* db,
                       const float* dg, const float* dbe, const float* dm,
                       const float* dv, char* ws) {
  const int tid = blockIdx.x * blockDim.x + threadIdx.x;
  _Float16* cdW = (_Float16*)(ws + OFF_CDW);
  _Float16* uW  = (_Float16*)(ws + OFF_UW);
  _Float16* dW  = (_Float16*)(ws + OFF_DW16);
  _Float16* pa  = (_Float16*)(ws + OFF_PA16);
  float* cdB  = (float*)(ws + OFF_CDB);
  float* c3B  = (float*)(ws + OFF_C3B);
  float* bAll = (float*)(ws + OFF_BALL);

  // Phase A: conv_down fold (L*64*64)
  if (tid < L_ * 64 * 64) {
    int c = tid & 63, o = (tid >> 6) & 63, i = tid >> 12;
    float s = cdg[i * 64 + o] * rsqrtf(cdv[i * 64 + o] + EPS_);
    cdW[tid] = (_Float16)(s * cdw[(i * 64 + o) * 64 + c]);
    if (c == 0)
      cdB[i * 64 + o] = s * (cdb[i * 64 + o] - cdm[i * 64 + o]) + cdbe[i * 64 + o];
  }
  // Phase B: per-layer 256-col weight (subsets folded with s-BN and final-BN
  // scale; c3 folded with final-BN scale).  (L*256*64)
  if (tid < L_ * 256 * 64) {
    int c = tid & 63, o = (tid >> 6) & 255, i = tid >> 14;
    float val;
    if (o < 192) {
      int j = o >> 6, oc = o & 63;
      float fs = bng[o] * rsqrtf(bnv[o] + EPS_);
      float ss = sg[(i * 3 + j) * 64 + oc] * rsqrtf(sv[(i * 3 + j) * 64 + oc] + EPS_);
      val = fs * ss * sw[((i * 3 + j) * 64 + oc) * 64 + c];
    } else {
      int cc = o - 192;
      float fs = bng[o] * rsqrtf(bnv[o] + EPS_);
      val = fs * c3w[(i * 64 + cc) * 64 + c];
      if (c == 0) c3B[i * 64 + cc] = fs * c3b[i * 64 + cc];
    }
    uW[tid] = (_Float16)val;
  }
  // Phase C: residual fold + combined per-channel bias (256*64)
  if (tid < 256 * 64) {
    int c = tid & 63, o = tid >> 6;
    float ds_ = dg[o] * rsqrtf(dv[o] + EPS_);
    dW[tid] = (_Float16)(ds_ * dw[o * 64 + c]);
    if (c == 0) {
      float fs  = bng[o] * rsqrtf(bnv[o] + EPS_);
      float fb  = bnb[o] - bnm[o] * fs;
      float res = ds_ * (db[o] - dm[o]) + dbe[o];
      float sbsum = 0.f;
      if (o < 192) {
        int j = o >> 6, oc = o & 63;
        for (int i = 0; i < L_; ++i) {
          float ss = sg[(i * 3 + j) * 64 + oc] * rsqrtf(sv[(i * 3 + j) * 64 + oc] + EPS_);
          sbsum += fs * (ss * (sb[(i * 3 + j) * 64 + oc] - sm[(i * 3 + j) * 64 + oc]) +
                         sbe[(i * 3 + j) * 64 + oc]);
        }
      }
      bAll[o] = res + fb + sbsum;
    }
  }
  // Phase D: PA padded 32x32 f16 (L*S*32*32)
  if (tid < L_ * S_ * 32 * 32) {
    int u = tid & 31, v = (tid >> 5) & 31, ij = tid >> 10;
    float val = (v < 25 && u < 25) ? PA[(ij * 25 + v) * 25 + u] : 0.f;
    pa[tid] = (_Float16)val;
  }
}

// ---------------------------------------------------------------------------
// K1: x (n,c,t,v) f32 -> xk f16 row-major-K, row=(n,t,v), K=c.
// ---------------------------------------------------------------------------
__global__ void k_xk(const float* x, char* ws) {
  int row = blockIdx.x * blockDim.x + threadIdx.x;
  if (row >= ROWS_) return;
  int n = row / (T_ * V_), rem = row % (T_ * V_), t = rem / V_, v = rem % V_;
  alignas(16) _Float16 h[64];
#pragma unroll
  for (int c = 0; c < 64; ++c)
    h[c] = (_Float16)x[(((size_t)n * 64 + c) * T_ + t) * V_ + v];
  float4* dst = (float4*)((_Float16*)(ws + OFF_XK) + (size_t)row * 64);
#pragma unroll
  for (int q = 0; q < 8; ++q) dst[q] = ((float4*)h)[q];
}

// ---------------------------------------------------------------------------
// K2: xd[i] = relu(BN(x @ cdw[i]^T)) for all 3 layers: GEMM 102400x64x192.
// ---------------------------------------------------------------------------
__global__ __launch_bounds__(256) void k_xd(char* ws) {
  const int lane = threadIdx.x & 31, lrow = lane & 15, lhalf = lane >> 4;
  const int w = threadIdx.x >> 5;
  const int m0 = blockIdx.x * 128 + w * 16;
  const _Float16* xk = (const _Float16*)(ws + OFF_XK);
  const _Float16* W  = (const _Float16*)(ws + OFF_CDW);
  const float*    Bv = (const float*)(ws + OFF_CDB);
  _Float16* xd = (_Float16*)(ws + OFF_XD);

  alignas(16) float4 rb[8];
  const float4* rp = (const float4*)(xk + (size_t)(m0 + lrow) * 64);
#pragma unroll
  for (int q = 0; q < 8; ++q) rb[q] = rp[q];
  half16 a0, a1;
  build_a((const _Float16*)rb, lhalf, a0, a1);

  f32x8 acc[12] = {};
#pragma unroll
  for (int nt = 0; nt < 12; ++nt) {
    int ocol = nt * 16 + lrow;
    alignas(16) float4 wb[8];
    const float4* wp = (const float4*)(W + (size_t)ocol * 64);
#pragma unroll
    for (int q = 0; q < 8; ++q) wb[q] = wp[q];
    half16 b0v, b1v;
    build_b((const _Float16*)wb, lhalf, b0v, b1v);
    acc[nt] = wmma16(a0, b0v, acc[nt]);
    acc[nt] = wmma16(a1, b1v, acc[nt]);
  }
#pragma unroll
  for (int nt = 0; nt < 12; ++nt) {
    int ocol = nt * 16 + lrow, i = ocol >> 6, oc = ocol & 63;
#pragma unroll
    for (int r = 0; r < 8; ++r) {
      int grow = m0 + r + (lhalf ? 8 : 0);
      float v = fmaxf(acc[nt][r] + Bv[ocol], 0.f);
      xd[(size_t)i * ROWS_ * 64 + (size_t)grow * 64 + oc] = (_Float16)v;
    }
  }
}

// ---------------------------------------------------------------------------
// K3: mean over T -> xdm[i][n][v][c] f32.
// ---------------------------------------------------------------------------
__global__ void k_mean(char* ws) {
  int tid = blockIdx.x * blockDim.x + threadIdx.x;
  if (tid >= L_ * N_ * V_ * 64) return;
  int c = tid & 63, v = (tid >> 6) % 25, n = (tid / (64 * 25)) % 32,
      i = tid / (64 * 25 * 32);
  const _Float16* xd = (const _Float16*)(ws + OFF_XD) + (size_t)i * ROWS_ * 64;
  float s = 0.f;
  for (int t = 0; t < T_; ++t)
    s += (float)xd[((size_t)((n * T_ + t) * V_ + v)) * 64 + c];
  ((float*)(ws + OFF_XDM))[((i * 32 + n) * 25 + v) * 64 + c] = s * (1.f / 128.f);
}

// ---------------------------------------------------------------------------
// K4: per (layer,n): x1/x2 (8x25), m[c][u][v] = c4 @ tanh(x1[u]-x2[v]) + c4b.
// ---------------------------------------------------------------------------
__global__ void k_m(const float* c1w, const float* c1b, const float* c2w,
                    const float* c2b, const float* c4w, const float* c4b,
                    char* ws) {
  int i = blockIdx.x >> 5, n = blockIdx.x & 31;
  __shared__ float xm[64][25];
  __shared__ float x1s[8][25], x2s[8][25];
  const float* xdm = (const float*)(ws + OFF_XDM);
  for (int idx = threadIdx.x; idx < 25 * 64; idx += blockDim.x) {
    int c = idx & 63, v = idx >> 6;
    xm[c][v] = xdm[((i * 32 + n) * 25 + v) * 64 + c];
  }
  __syncthreads();
  for (int idx = threadIdx.x; idx < 400; idx += blockDim.x) {
    int half = idx / 200, rem = idx % 200, r = rem / 25, v = rem % 25;
    const float* wp = (half ? c2w : c1w) + (i * 8 + r) * 64;
    float s = (half ? c2b : c1b)[i * 8 + r];
    for (int c = 0; c < 64; ++c) s += wp[c] * xm[c][v];
    if (half) x2s[r][v] = s; else x1s[r][v] = s;
  }
  __syncthreads();
  _Float16* mb = (_Float16*)(ws + OFF_MBUF);
  for (int idx = threadIdx.x; idx < 64 * 625; idx += blockDim.x) {
    int c = idx / 625, rem = idx % 625, u = rem / 25, v = rem % 25;
    float s = c4b[i * 64 + c];
#pragma unroll
    for (int r = 0; r < 8; ++r)
      s += c4w[(i * 64 + c) * 8 + r] * tanhf(x1s[r][u] - x2s[r][v]);
    mb[((size_t)((i * 32 + n) * 64 + c)) * 625 + rem] = (_Float16)s;
  }
}

// ---------------------------------------------------------------------------
// K5: per layer: GEMM 102400x64x256.
//     cols 0..191 (subset convs)  -> U   (row-major, stride 192)
//     cols 192..255 (c3 conv)     -> x3T (transposed [l][n][c][t*25+v], +bias)
// ---------------------------------------------------------------------------
__global__ __launch_bounds__(256) void k_U(char* ws) {
  const int lane = threadIdx.x & 31, lrow = lane & 15, lhalf = lane >> 4;
  const int w = threadIdx.x >> 5, i = blockIdx.z;
  const int m0 = blockIdx.x * 128 + w * 16;
  const _Float16* xd = (const _Float16*)(ws + OFF_XD) + (size_t)i * ROWS_ * 64;
  const _Float16* W  = (const _Float16*)(ws + OFF_UW) + (size_t)i * 256 * 64;
  const float* c3B = (const float*)(ws + OFF_C3B);
  _Float16* U   = (_Float16*)(ws + OFF_U)   + (size_t)i * ROWS_ * 192;
  _Float16* x3T = (_Float16*)(ws + OFF_X3T) + (size_t)i * N_ * 64 * (T_ * V_);

  alignas(16) float4 rb[8];
  const float4* rp = (const float4*)(xd + (size_t)(m0 + lrow) * 64);
#pragma unroll
  for (int q = 0; q < 8; ++q) rb[q] = rp[q];
  half16 a0, a1;
  build_a((const _Float16*)rb, lhalf, a0, a1);

  f32x8 acc[16] = {};
#pragma unroll
  for (int nt = 0; nt < 16; ++nt) {
    int ocol = nt * 16 + lrow;
    alignas(16) float4 wb[8];
    const float4* wp = (const float4*)(W + (size_t)ocol * 64);
#pragma unroll
    for (int q = 0; q < 8; ++q) wb[q] = wp[q];
    half16 b0v, b1v;
    build_b((const _Float16*)wb, lhalf, b0v, b1v);
    acc[nt] = wmma16(a0, b0v, acc[nt]);
    acc[nt] = wmma16(a1, b1v, acc[nt]);
  }
#pragma unroll
  for (int nt = 0; nt < 16; ++nt) {
    int ocol = nt * 16 + lrow;
    if (ocol < 192) {
#pragma unroll
      for (int r = 0; r < 8; ++r) {
        int grow = m0 + r + (lhalf ? 8 : 0);
        U[(size_t)grow * 192 + ocol] = (_Float16)acc[nt][r];
      }
    } else {
      int cc = ocol - 192;
      float bias = c3B[i * 64 + cc];
#pragma unroll
      for (int r = 0; r < 8; ++r) {
        int grow = m0 + r + (lhalf ? 8 : 0);
        int n = grow / (T_ * V_), rem = grow % (T_ * V_);
        x3T[((size_t)n * 64 + cc) * (T_ * V_) + rem] = (_Float16)(acc[nt][r] + bias);
      }
    }
  }
}

// ---------------------------------------------------------------------------
// K6: CTRGC depthwise einsum as WMMA per (n,c):
//     cbuf[n,c,t,u] = sum_i sum_v x3_i[t,v] * m_i[u,v].
//     x3 loaded coalesced from x3T; fragments packed from b128 LDS rows.
// ---------------------------------------------------------------------------
__global__ __launch_bounds__(256) void k_ctr(char* ws) {
  int n = blockIdx.x >> 6, c = blockIdx.x & 63;
  __shared__ _Float16 x3s[3][128][32];
  __shared__ _Float16 ms[3][32][32];
  const _Float16* x3T = (const _Float16*)(ws + OFF_X3T);
  const _Float16* mb  = (const _Float16*)(ws + OFF_MBUF);

  // coalesced b128 loads of x3 (contiguous per (i,n,c)), scatter into padded LDS
  for (int idx = threadIdx.x; idx < 3 * 400; idx += blockDim.x) {
    int i = idx / 400, f4 = idx % 400;
    float4 tmp = ((const float4*)(x3T + ((size_t)i * N_ * 64 + (size_t)n * 64 + c) *
                                            (T_ * V_)))[f4];
    const _Float16* hp = (const _Float16*)&tmp;
#pragma unroll
    for (int e = 0; e < 8; ++e) {
      int flat = f4 * 8 + e, t = flat / 25, vv = flat % 25;
      x3s[i][t][vv] = hp[e];
    }
  }
  // zero the v-pad columns
  for (int idx = threadIdx.x; idx < 3 * 128 * 7; idx += blockDim.x) {
    int i = idx / (128 * 7), rem = idx % (128 * 7), t = rem / 7, vv = 25 + rem % 7;
    x3s[i][t][vv] = (_Float16)0.f;
  }
  for (int idx = threadIdx.x; idx < 3 * 32 * 32; idx += blockDim.x) {
    int i = idx >> 10, rem = idx & 1023, u = rem >> 5, vv = rem & 31;
    _Float16 val = (_Float16)0.f;
    if (u < 25 && vv < 25)
      val = mb[((size_t)((i * 32 + n) * 64 + c)) * 625 + u * 25 + vv];
    ms[i][u][vv] = val;
  }
  __syncthreads();
  const int lane = threadIdx.x & 31, lrow = lane & 15, lhalf = lane >> 4;
  const int w = threadIdx.x >> 5, t0 = w * 16;
  f32x8 acc0 = {}, acc1 = {};
#pragma unroll
  for (int i = 0; i < 3; ++i) {
    alignas(16) float4 xr[4], m0r[4], m1r[4];
    const float4* xp  = (const float4*)&x3s[i][t0 + lrow][0];
    const float4* mp0 = (const float4*)&ms[i][lrow][0];
    const float4* mp1 = (const float4*)&ms[i][16 + lrow][0];
#pragma unroll
    for (int q = 0; q < 4; ++q) { xr[q] = xp[q]; m0r[q] = mp0[q]; m1r[q] = mp1[q]; }
    half16 a   = build_a32((const _Float16*)xr, lhalf);
    half16 bm0 = build_b32((const _Float16*)m0r, lhalf);
    half16 bm1 = build_b32((const _Float16*)m1r, lhalf);
    acc0 = wmma16(a, bm0, acc0);
    acc1 = wmma16(a, bm1, acc1);
  }
  float* cb = (float*)(ws + OFF_CBUF);
#pragma unroll
  for (int ut = 0; ut < 2; ++ut) {
    f32x8 A = ut ? acc1 : acc0;
    int u = ut * 16 + lrow;
    if (u < 25)
#pragma unroll
      for (int r = 0; r < 8; ++r) {
        int t = t0 + r + (lhalf ? 8 : 0);
        cb[((size_t)(n * 64 + c) * T_ + t) * V_ + u] = A[r];
      }
  }
}

// ---------------------------------------------------------------------------
// K7: final: residual GEMM (LDS-staged x tile) + PA v-mix WMMA over
//     LDS-staged U slices + CTRGC add + combined bias + ReLU.
//     grid: (n:32) x (t-tile:8) x (channel quarter q:4)
//     LDS: one 51.2KB buffer aliased between xA (residual) and Ut (v-mix).
// ---------------------------------------------------------------------------
__global__ __launch_bounds__(256) void k_final(char* ws, float* out) {
  const int bx = blockIdx.x;
  const int n = bx >> 5, tt = (bx >> 2) & 7, q = bx & 3;
  const int t0 = tt * 16;
  const int lane = threadIdx.x & 31, lrow = lane & 15, lhalf = lane >> 4;
  const int w = threadIdx.x >> 5;

  __shared__ alignas(16) _Float16 smem[400 * 64];  // 51.2 KB, aliased
  const _Float16* xk  = (const _Float16*)(ws + OFF_XK);
  const _Float16* Uh  = (const _Float16*)(ws + OFF_U);
  const _Float16* dW  = (const _Float16*)(ws + OFF_DW16);
  const _Float16* pa  = (const _Float16*)(ws + OFF_PA16);
  const float* bAll   = (const float*)(ws + OFF_BALL);
  const float* cb     = (const float*)(ws + OFF_CBUF);

  // ---- phase 1: residual GEMM from x tile ----
  for (int idx = threadIdx.x; idx < 400 * 8; idx += blockDim.x) {
    int rr = idx >> 3, qq = idx & 7;
    int tl = rr / 25, v = rr % 25;
    ((float4*)smem)[rr * 8 + qq] =
        ((const float4*)(xk + ((size_t)((n * T_ + t0 + tl) * V_ + v)) * 64))[qq];
  }
  __syncthreads();

  half16 bd0[4], bd1[4];
#pragma unroll
  for (int nt = 0; nt < 4; ++nt) {
    int ocol = q * 64 + nt * 16 + lrow;
    alignas(16) float4 wb[8];
    const float4* wp = (const float4*)(dW + (size_t)ocol * 64);
#pragma unroll
    for (int qq = 0; qq < 8; ++qq) wb[qq] = wp[qq];
    build_b((const _Float16*)wb, lhalf, bd0[nt], bd1[nt]);
  }

  f32x8 acc[2][2][4] = {};  // [tsel][vhalf][ntile]

#pragma unroll
  for (int ts = 0; ts < 2; ++ts) {
    int tl = w + ts * 8;
#pragma unroll
    for (int vh = 0; vh < 2; ++vh) {
      int v = vh * 16 + lrow;
      half16 a0 = {}, a1 = {};
      if (v < 25) {
        alignas(16) float4 rb[8];
        const float4* rp = (const float4*)&smem[(tl * 25 + v) * 64];
#pragma unroll
        for (int qq = 0; qq < 8; ++qq) rb[qq] = rp[qq];
        build_a((const _Float16*)rb, lhalf, a0, a1);
      }
#pragma unroll
      for (int nt = 0; nt < 4; ++nt) {
        acc[ts][vh][nt] = wmma16(a0, bd0[nt], acc[ts][vh][nt]);
        acc[ts][vh][nt] = wmma16(a1, bd1[nt], acc[ts][vh][nt]);
      }
    }
  }
  __syncthreads();  // xA no longer needed; smem becomes the U staging buffer

  // ---- phase 2 (subset quarters): PA v-mix over LDS-staged U slices ----
  // out[v][o] += sum_i sum_u PA[i,q][v,u] * U_i[t,u][o]
  if (q < 3) {
    for (int i = 0; i < 3; ++i) {
      // stage U_i slice: Ut[tloc(16)][u(25)][ocol(64)]  = 51.2 KB, coalesced
      const _Float16* Ubase = Uh + (size_t)i * ROWS_ * 192;
      for (int idx = threadIdx.x; idx < 400 * 8; idx += blockDim.x) {
        int rr = idx >> 3, qq = idx & 7;
        int tl = rr / 25, u = rr % 25;
        const float4* src = (const float4*)(
            Ubase + ((size_t)((n * T_ + t0 + tl) * V_ + u)) * 192 + q * 64);
        ((float4*)smem)[rr * 8 + qq] = src[qq];
        if (i + 1 < 3)  // warm next layer's slice in L2 (global_prefetch_b8)
          __builtin_prefetch(
              Ubase + (size_t)ROWS_ * 192 +
                  ((size_t)((n * T_ + t0 + tl) * V_ + u)) * 192 + q * 64 + qq * 8,
              0, 2);
      }
      __syncthreads();

      half16 apa[2];
#pragma unroll
      for (int vh = 0; vh < 2; ++vh) {
        alignas(16) float4 pr[4];
        const float4* pp = (const float4*)(
            pa + ((size_t)((i * 3 + q) * 32 + vh * 16 + lrow)) * 32);
#pragma unroll
        for (int qq = 0; qq < 4; ++qq) pr[qq] = pp[qq];
        apa[vh] = build_a32((const _Float16*)pr, lhalf);
      }
#pragma unroll
      for (int ts = 0; ts < 2; ++ts) {
        int tl = w + ts * 8;
#pragma unroll
        for (int nt = 0; nt < 4; ++nt) {
          int ocl = nt * 16 + lrow;
          half16 bu = {};
#pragma unroll
          for (int vg = 0; vg < 8; ++vg)
#pragma unroll
            for (int e = 0; e < 2; ++e) {
              int u = vg * 2 + e + (lhalf ? 16 : 0);
              _Float16 val = (_Float16)0.f;
              if (u < 25) val = smem[(tl * 25 + u) * 64 + ocl];
              bu[vg * 2 + e] = val;
            }
#pragma unroll
          for (int vh = 0; vh < 2; ++vh)
            acc[ts][vh][nt] = wmma16(apa[vh], bu, acc[ts][vh][nt]);
        }
      }
      __syncthreads();  // before restaging smem with the next layer
    }
  }

  // ---- epilogue: CTRGC add (q==3) + bias + ReLU + store ----
#pragma unroll
  for (int ts = 0; ts < 2; ++ts) {
    int t = t0 + w + ts * 8;
#pragma unroll
    for (int vh = 0; vh < 2; ++vh)
#pragma unroll
      for (int nt = 0; nt < 4; ++nt) {
        int ocol = q * 64 + nt * 16 + lrow;
#pragma unroll
        for (int r = 0; r < 8; ++r) {
          int v = vh * 16 + r + (lhalf ? 8 : 0);
          if (v < 25) {
            float val = acc[ts][vh][nt][r];
            if (q == 3)
              val += cb[((size_t)(n * 64 + nt * 16 + lrow) * T_ + t) * V_ + v];
            val += bAll[ocol];
            val = fmaxf(val, 0.f);
            out[((size_t)(n * 256 + ocol) * T_ + t) * V_ + v] = val;
          }
        }
      }
  }
}

// ---------------------------------------------------------------------------
extern "C" void kernel_launch(void* const* d_in, const int* in_sizes, int n_in,
                              void* d_out, int out_size, void* d_ws,
                              size_t ws_size, hipStream_t stream) {
  const float* x    = (const float*)d_in[0];
  const float* PA   = (const float*)d_in[1];
  const float* cdw  = (const float*)d_in[2];
  const float* cdb  = (const float*)d_in[3];
  const float* cdg  = (const float*)d_in[4];
  const float* cdbe = (const float*)d_in[5];
  const float* cdm  = (const float*)d_in[6];
  const float* cdv  = (const float*)d_in[7];
  const float* sw   = (const float*)d_in[8];
  const float* sb   = (const float*)d_in[9];
  const float* sg   = (const float*)d_in[10];
  const float* sbe  = (const float*)d_in[11];
  const float* sm   = (const float*)d_in[12];
  const float* sv   = (const float*)d_in[13];
  const float* c1w  = (const float*)d_in[14];
  const float* c1b  = (const float*)d_in[15];
  const float* c2w  = (const float*)d_in[16];
  const float* c2b  = (const float*)d_in[17];
  const float* c3w  = (const float*)d_in[18];
  const float* c3b  = (const float*)d_in[19];
  const float* c4w  = (const float*)d_in[20];
  const float* c4b  = (const float*)d_in[21];
  const float* bng  = (const float*)d_in[22];
  const float* bnb  = (const float*)d_in[23];
  const float* bnm  = (const float*)d_in[24];
  const float* bnv  = (const float*)d_in[25];
  const float* dw   = (const float*)d_in[26];
  const float* db   = (const float*)d_in[27];
  const float* dg   = (const float*)d_in[28];
  const float* dbe  = (const float*)d_in[29];
  const float* dm   = (const float*)d_in[30];
  const float* dv   = (const float*)d_in[31];
  char* ws = (char*)d_ws;
  float* out = (float*)d_out;

  k_prep<<<192, 256, 0, stream>>>(PA, cdw, cdb, cdg, cdbe, cdm, cdv, sw, sb, sg,
                                  sbe, sm, sv, c3w, c3b, bng, bnb, bnm, bnv, dw,
                                  db, dg, dbe, dm, dv, ws);
  k_xk<<<ROWS_ / 256, 256, 0, stream>>>(x, ws);
  k_xd<<<ROWS_ / 128, 256, 0, stream>>>(ws);
  k_mean<<<(L_ * N_ * V_ * 64 + 255) / 256, 256, 0, stream>>>(ws);
  k_m<<<L_ * N_, 256, 0, stream>>>(c1w, c1b, c2w, c2b, c4w, c4b, ws);
  k_U<<<dim3(ROWS_ / 128, 1, L_), 256, 0, stream>>>(ws);
  k_ctr<<<N_ * 64, 256, 0, stream>>>(ws);
  k_final<<<N_ * 8 * 4, 256, 0, stream>>>(ws, out);
}
